// AnnRC_1864015807039
// MI455X (gfx1250) — matrix operand order repaired and checked by reading, not compiled, not run on hardware
//
#include <hip/hip_runtime.h>
#include <hip/hip_bf16.h>
#include <math.h>

// Problem sizes (fixed by the reference)
#define BATCH   256
#define FRAMES  128
#define NIN     700
#define NINP    704              // NIN padded to a multiple of 32 (zeros)
#define NHID    2048
#define MALL    (BATCH*FRAMES)   // 32768 rows of the hoisted feed-forward GEMM
#define KT_IN   (NINP/32)        // 22 K-tiles for x @ W_in
#define KT_H    (NHID/32)        // 64 K-tiles for spike @ A

typedef __attribute__((ext_vector_type(16))) __bf16       v16bf;
typedef __attribute__((ext_vector_type(8)))  float        v8f;
typedef __attribute__((ext_vector_type(8)))  unsigned int v8u;
typedef __attribute__((ext_vector_type(4)))  unsigned int v4u;
typedef __attribute__((ext_vector_type(8)))  int          v8i_;
typedef __attribute__((ext_vector_type(4)))  int          v4i_;

// TDM path only on the device pass when the builtin exists; otherwise fall back
// to the (already working) direct global-load path.
#if defined(__HIP_DEVICE_COMPILE__) && defined(__gfx1250__) && \
    __has_builtin(__builtin_amdgcn_tensor_load_to_lds)
#define HAVE_TDM 1
#else
#define HAVE_TDM 0
#endif

union BF16Op { v8u u; v16bf b; };

__device__ __forceinline__ unsigned short f32_to_bf16(float f) {
  unsigned int u = __float_as_uint(f);
  u += 0x7fffu + ((u >> 16) & 1u);          // round-to-nearest-even
  return (unsigned short)(u >> 16);
}
__device__ __forceinline__ float bf16_to_f32(unsigned short h) {
  return __uint_as_float(((unsigned int)h) << 16);
}

// u16 index of element (n,k) inside packed WMMA B-operand tiles (32K x 16N, bf16)
// Layout: VGPR 0..3 -> K 0..7 (+8 for lanes 16-31), VGPR 4..7 -> K 16..23 (+8), N = lane&15.
__device__ __forceinline__ size_t b_pack_idx(int n, int k, int ktiles) {
  int tile = (n >> 4) * ktiles + (k >> 5);
  int lane = (n & 15) | (((k >> 3) & 1) << 4);
  int j    = (((k >> 4) & 1) << 2) | ((k & 7) >> 1);
  return ((((size_t)tile * 32 + lane) * 8 + j) << 1) | (size_t)(k & 1);
}
// u16 index of element (m,k) inside packed WMMA A-operand tiles (16M x 32K, bf16)
// Layout per ISA 7.12.2: VGPR 0..3 -> K 0..7 (+8 for lanes 16-31), VGPR 4..7 -> K 16..23 (+8).
__device__ __forceinline__ size_t a_pack_idx(int m, int k, int ktiles) {
  int tile = (m >> 4) * ktiles + (k >> 5);
  int lane = (m & 15) | (((k >> 3) & 1) << 4);
  int j    = (((k >> 4) & 1) << 2) | ((k & 7) >> 1);
  return ((((size_t)tile * 32 + lane) * 8 + j) << 1) | (size_t)(k & 1);
}

// Each lane reads its 32 contiguous bytes of a packed operand tile (2x global_load_b128).
__device__ __forceinline__ v16bf load_op(const unsigned short* __restrict__ base,
                                         size_t tile, int lane) {
  BF16Op t;
  t.u = *(const v8u*)(base + tile * 512 + (size_t)lane * 16);
  return t.b;
}

__device__ __forceinline__ v8f wmma_bf16(v16bf a, v16bf b, v8f c) {
  return __builtin_amdgcn_wmma_f32_16x16x32_bf16(false, a, false, b, (short)0, c,
                                                 false, false);
}

#if HAVE_TDM
// TDM 2-D copy: `rows` rows of `row_qwords` 8-byte elements, global row stride
// `stride_qwords` (8B units), packed contiguously into LDS at byte offset lds_off.
// D# per CDNA5 ISA 8.3/8.4 (groups 2/3 zero => <=2-D tensor).
// This toolchain's builtin is the 6-arg form:
//   (uint32x4 g0, int32x8 g1, int32x4 g2, int32x4 g3, int32x8 extra, i32 cpol)
__device__ __forceinline__ void tdm_copy_2d(const void* gsrc, unsigned lds_off,
                                            unsigned rows, unsigned row_qwords,
                                            unsigned stride_qwords) {
  unsigned long long ga = (unsigned long long)(size_t)gsrc;
  v4u g0;
  g0.x = 1u;                                   // count=1 user descriptor, no gather
  g0.y = lds_off;                              // lds_addr (bytes)
  g0.z = (unsigned)ga;                         // global_addr[31:0]
  g0.w = (unsigned)(ga >> 32) | (2u << 30);    // global_addr[56:32] | type=2 ("image")
  v8i_ g1;
  g1[0] = (int)(3u << 16);                     // data_size = 8 bytes; no multicast/pad
  g1[1] = (int)(row_qwords << 16);             // tensor_dim0[15:0]
  g1[2] = (int)(rows << 16);                   // tensor_dim1[15:0] (tensor_dim0 hi = 0)
  g1[3] = (int)(row_qwords << 16);             // tile_dim0
  g1[4] = (int)rows;                           // tile_dim1 (tile_dim2 = 0)
  g1[5] = (int)stride_qwords;                  // tensor_dim0_stride[31:0]
  g1[6] = 0;                                   // stride hi / tensor_dim1_stride lo
  g1[7] = 0;                                   // tensor_dim1_stride hi
  v4i_ z4 = {0, 0, 0, 0};
  v8i_ z8 = {0, 0, 0, 0, 0, 0, 0, 0};
  __builtin_amdgcn_tensor_load_to_lds(g0, g1, z4, z4, z8, 0);
}

// Fetch this block's 4 B-operand N-tiles (hi + lo) for k-step kk into LDS buffer.
__device__ __forceinline__ void tdm_fetch_B(const unsigned short* __restrict__ ahi,
                                            const unsigned short* __restrict__ alo,
                                            int nt0, int kk, unsigned lds_off) {
  size_t t0 = (size_t)nt0 * KT_H + kk;         // first packed tile (1 KB each)
  tdm_copy_2d(ahi + t0 * 512, lds_off,        4u, 128u, (unsigned)KT_H * 128u);
  tdm_copy_2d(alo + t0 * 512, lds_off + 4096, 4u, 128u, (unsigned)KT_H * 128u);
}
#endif

// ---------------- packing / init kernels (run once per launch; bandwidth-trivial) ----

// A[k][n] (f32, row-major 2048x2048) -> split bf16 hi/lo in packed B-operand layout
__global__ void pack_Anet(const float* __restrict__ A,
                          unsigned short* __restrict__ hi,
                          unsigned short* __restrict__ lo) {
  size_t i = (size_t)blockIdx.x * blockDim.x + threadIdx.x;
  if (i >= (size_t)NHID * NHID) return;
  int k = (int)(i >> 11);
  int n = (int)(i & (NHID - 1));
  float v = A[i];
  unsigned short h = f32_to_bf16(v);
  unsigned short l = f32_to_bf16(v - bf16_to_f32(h));
  size_t idx = b_pack_idx(n, k, KT_H);
  hi[idx] = h;
  lo[idx] = l;
}

// W_in[k][n] (f32, 700x2048) -> split bf16 hi/lo packed B-operand tiles, K padded to 704
__global__ void pack_Win(const float* __restrict__ W,
                         unsigned short* __restrict__ hi,
                         unsigned short* __restrict__ lo) {
  size_t i = (size_t)blockIdx.x * blockDim.x + threadIdx.x;
  if (i >= (size_t)NINP * NHID) return;
  int k = (int)(i >> 11);
  int n = (int)(i & (NHID - 1));
  float v = (k < NIN) ? W[(size_t)k * NHID + n] : 0.0f;
  unsigned short h = f32_to_bf16(v);
  unsigned short l = f32_to_bf16(v - bf16_to_f32(h));
  size_t idx = b_pack_idx(n, k, KT_IN);
  hi[idx] = h;
  lo[idx] = l;
}

// x[m][k] (f32, 32768x700) -> split bf16 hi/lo packed A-operand tiles, K padded to 704
__global__ void pack_x(const float* __restrict__ x,
                       unsigned short* __restrict__ hi,
                       unsigned short* __restrict__ lo) {
  unsigned int i = blockIdx.x * blockDim.x + threadIdx.x;
  if (i >= (unsigned int)MALL * NINP) return;
  unsigned int m = i / NINP;
  unsigned int k = i - m * NINP;
  float v = (k < NIN) ? x[(size_t)m * NIN + k] : 0.0f;
  unsigned short h = f32_to_bf16(v);
  unsigned short l = f32_to_bf16(v - bf16_to_f32(h));
  size_t idx = a_pack_idx((int)m, (int)k, KT_IN);
  hi[idx] = h;
  lo[idx] = l;
}

// mem := mem_init, spike buffer 0 := 0 (packed A-operand layout)
__global__ void init_state(const float* __restrict__ mem_init,
                           float* __restrict__ mem,
                           unsigned short* __restrict__ s0) {
  size_t i = (size_t)blockIdx.x * blockDim.x + threadIdx.x;
  if (i >= (size_t)BATCH * NHID) return;
  int b = (int)(i >> 11);
  int n = (int)(i & (NHID - 1));
  mem[i] = mem_init[i];
  s0[a_pack_idx(b, n, KT_H)] = 0;
}

// ---------------- hoisted feed-forward GEMM: U = x @ W_in (3-term bf16 split) -------
// grid (32, 512), 128 threads (4 waves); wave tile = 16M x 64N, K = 704.
__global__ __launch_bounds__(128) void ugemm_kernel(
    const unsigned short* __restrict__ xhi, const unsigned short* __restrict__ xlo,
    const unsigned short* __restrict__ whi, const unsigned short* __restrict__ wlo,
    float* __restrict__ U) {
  int lane  = threadIdx.x & 31;
  int wave  = threadIdx.x >> 5;
  int mtile = blockIdx.y * 4 + wave;   // 0..2047
  int nt0   = blockIdx.x * 4;          // 0..124 step 4
  v8f acc[4] = {};
  for (int kk = 0; kk < KT_IN; ++kk) {
    v16bf ah = load_op(xhi, (size_t)mtile * KT_IN + kk, lane);
    v16bf al = load_op(xlo, (size_t)mtile * KT_IN + kk, lane);
#pragma unroll
    for (int nt = 0; nt < 4; ++nt) {
      v16bf bh = load_op(whi, (size_t)(nt0 + nt) * KT_IN + kk, lane);
      v16bf bl = load_op(wlo, (size_t)(nt0 + nt) * KT_IN + kk, lane);
      acc[nt] = wmma_bf16(ah, bh, acc[nt]);   // hi*hi
      acc[nt] = wmma_bf16(ah, bl, acc[nt]);   // hi*lo
      acc[nt] = wmma_bf16(al, bh, acc[nt]);   // lo*hi
    }
  }
  int hv = lane >> 4, c = lane & 15;
#pragma unroll
  for (int nt = 0; nt < 4; ++nt) {
    int n = (nt0 + nt) * 16 + c;
#pragma unroll
    for (int j = 0; j < 8; ++j) {
      int m = mtile * 16 + j + 8 * hv;   // C/D layout: VGPR j -> M=j (+8 for lanes 16-31)
      U[(size_t)m * NHID + n] = acc[nt][j];
    }
  }
}

// ---------------- one recurrent frame: r = spike@A (2-term EXACT split since spike
// is binary), then fused tanh / membrane / threshold update --------------------------
// grid (32, 2), 256 threads (8 waves); wave tile = 16M x 64N, K = 2048.
// All 8 waves of a block share the same 4 B-tiles per k-step, so wave 0 DMAs them
// into a double-buffered LDS tile via the TDM; compute waves read ds_load_b128.
__global__ __launch_bounds__(256) void step_kernel(
    const unsigned short* __restrict__ ahi, const unsigned short* __restrict__ alo,
    const unsigned short* __restrict__ spike_in, unsigned short* __restrict__ spike_out,
    const float* __restrict__ U, const float* __restrict__ bias,
    float* __restrict__ mem, float* __restrict__ mems_out,
    float* __restrict__ spikes_out, int t) {
  int lane  = threadIdx.x & 31;
  int wave  = threadIdx.x >> 5;
  int mtile = blockIdx.y * 8 + wave;   // 0..15
  int nt0   = blockIdx.x * 4;
  v8f acc[4] = {};

#if HAVE_TDM
  __shared__ v8u lbuf[512];            // 2 x (4 KB hi + 4 KB lo) double buffer
  bool leader = (wave == 0);
  unsigned lds_base = (unsigned)(size_t)&lbuf[0];   // low 32 bits = LDS byte offset
  if (leader) tdm_fetch_B(ahi, alo, nt0, 0, lds_base);
#endif

  for (int kk = 0; kk < KT_H; ++kk) {
#if HAVE_TDM
    if (leader) __builtin_amdgcn_s_wait_tensorcnt(0);   // buffer kk&1 landed
    __syncthreads();                                    // visible to all 8 waves
    if (leader && (kk + 1) < KT_H)                      // prefetch next k-step
      tdm_fetch_B(ahi, alo, nt0, kk + 1, lds_base + (unsigned)((kk + 1) & 1) * 8192u);
    int lb = (kk & 1) * 256;                            // v8u index of current buffer
#endif
    v16bf s = load_op(spike_in, (size_t)mtile * KT_H + kk, lane);
#pragma unroll
    for (int nt = 0; nt < 4; ++nt) {
#if HAVE_TDM
      BF16Op th, tl;
      th.u = lbuf[lb + nt * 32 + lane];         // ds_load_b128 x2 (hi)
      tl.u = lbuf[lb + 128 + nt * 32 + lane];   // ds_load_b128 x2 (lo)
      v16bf bh = th.b, bl = tl.b;
#else
      v16bf bh = load_op(ahi, (size_t)(nt0 + nt) * KT_H + kk, lane);
      v16bf bl = load_op(alo, (size_t)(nt0 + nt) * KT_H + kk, lane);
#endif
      acc[nt] = wmma_bf16(s, bh, acc[nt]);
      acc[nt] = wmma_bf16(s, bl, acc[nt]);
    }
  }

  int hv = lane >> 4, c = lane & 15;
#pragma unroll
  for (int nt = 0; nt < 4; ++nt) {
    int n = (nt0 + nt) * 16 + c;
    float bn_bias = bias[n];
#pragma unroll
    for (int j = 0; j < 8; ++j) {
      int b = mtile * 16 + j + 8 * hv;
      size_t bn  = (size_t)b * NHID + n;
      size_t btn = ((size_t)b * FRAMES + t) * NHID + n;
      float r = acc[nt][j];
      float u = U[btn];
      float y = tanhf(0.5f * r + 0.5f * (u + bn_bias));
      size_t sidx = a_pack_idx(b, n, KT_H);
      float sp_old = spike_in[sidx] ? 1.0f : 0.0f;   // bf16 1.0 == 0x3F80
      float m_new  = mem[bn] * 0.5f - 0.5f * (1.0f - sp_old) + y;
      float sp_new = (m_new > 0.5f) ? 1.0f : 0.0f;
      mem[bn]         = m_new;
      mems_out[btn]   = m_new;
      spikes_out[btn] = sp_new;
      spike_out[sidx] = (m_new > 0.5f) ? (unsigned short)0x3F80 : (unsigned short)0;
    }
  }
}

extern "C" void kernel_launch(void* const* d_in, const int* in_sizes, int n_in,
                              void* d_out, int out_size, void* d_ws, size_t ws_size,
                              hipStream_t stream) {
  const float* x        = (const float*)d_in[0];   // [256,128,700]
  const float* W_in     = (const float*)d_in[1];   // [700,2048]
  const float* A        = (const float*)d_in[2];   // [2048,2048]
  const float* bias     = (const float*)d_in[3];   // [2048]
  const float* mem_init = (const float*)d_in[4];   // [256,2048]

  float* mems   = (float*)d_out;                               // [256,128,2048]
  float* spikes = mems + (size_t)MALL * NHID;                  // [256,128,2048]

  // Workspace carve-up (~370 MB; all regions fully rewritten every launch).
  char* ws = (char*)d_ws;
  size_t o = 0;
  float*          U   = (float*)(ws + o);          o += (size_t)MALL * NHID * 4;  // 268 MB
  unsigned short* Ahi = (unsigned short*)(ws + o); o += (size_t)NHID * NHID * 2;  // 8 MB
  unsigned short* Alo = (unsigned short*)(ws + o); o += (size_t)NHID * NHID * 2;  // 8 MB
  unsigned short* Whi = (unsigned short*)(ws + o); o += (size_t)NINP * NHID * 2;
  unsigned short* Wlo = (unsigned short*)(ws + o); o += (size_t)NINP * NHID * 2;
  unsigned short* Xhi = (unsigned short*)(ws + o); o += (size_t)MALL * NINP * 2;  // 46 MB
  unsigned short* Xlo = (unsigned short*)(ws + o); o += (size_t)MALL * NINP * 2;  // 46 MB
  unsigned short* S0  = (unsigned short*)(ws + o); o += (size_t)BATCH * NHID * 2;
  unsigned short* S1  = (unsigned short*)(ws + o); o += (size_t)BATCH * NHID * 2;
  float*          Mem = (float*)(ws + o);          o += (size_t)BATCH * NHID * 4;
  (void)o; (void)ws_size; (void)in_sizes; (void)n_in; (void)out_size;

  // One-time (per launch) operand packing + state init.
  pack_Anet<<<(NHID * NHID) / 256, 256, 0, stream>>>(A, Ahi, Alo);
  pack_Win<<<(NINP * NHID) / 256, 256, 0, stream>>>(W_in, Whi, Wlo);
  pack_x<<<((unsigned)MALL * NINP) / 256, 256, 0, stream>>>(x, Xhi, Xlo);
  init_state<<<(BATCH * NHID) / 256, 256, 0, stream>>>(mem_init, Mem, S0);

  // Hoisted feed-forward GEMM for all 128 frames at once.
  ugemm_kernel<<<dim3(32, 512), 128, 0, stream>>>(Xhi, Xlo, Whi, Wlo, U);

  // 128 sequential recurrent frames; spike double-buffered across launches.
  for (int t = 0; t < FRAMES; ++t) {
    const unsigned short* sin = (t & 1) ? S1 : S0;
    unsigned short* sout      = (t & 1) ? S0 : S1;
    step_kernel<<<dim3(32, 2), 256, 0, stream>>>(Ahi, Alo, sin, sout, U, bias,
                                                 Mem, mems, spikes, t);
  }
}
